// LFSA_3152505995519
// MI455X (gfx1250) — matrix operand order repaired and checked by reading, not compiled, not run on hardware
//
#include <hip/hip_runtime.h>
#include <hip/hip_bf16.h>

typedef __attribute__((ext_vector_type(16))) _Float16 v16h;
typedef __attribute__((ext_vector_type(8)))  float    v8f;

#define DIMC 64
#define HH 256
#define WW 256
#define HALO 10          // 8x8 patch + 1-px halo for depthwise 3x3
#define NPIX 100         // 10*10
#define NPAD 112         // padded to 7 N-tiles of 16 for WMMA

// LDS partition offsets (bytes)
#define OFF_XH   0                        // _Float16 xh[NPAD][64]      14336 B
#define OFF_WHS  14336                    // _Float16 whs[48][64]        6144 B
#define OFF_HID  (OFF_WHS+6144)           // float    hid[48][NPAD]     21504 B
#define OFF_QB   (OFF_HID+21504)          // float    qb[16][64]         4096 B
#define OFF_KB   (OFF_QB+4096)            // float    kb[16][64]         4096 B
#define OFF_OB   (OFF_KB+4096)            // float    ob[128][64]       32768 B
#define OFF_VB   (OFF_OB+32768)           // float    vb[128][64]       32768 B
#define OFF_GB   (OFF_VB+32768)           // _Float16 gb[64][128]       16384 B
#define OFF_MU   (OFF_GB+16384)           // float    mu[64]              256 B
#define OFF_RS   (OFF_MU+256)             // float    rstd[64]            256 B
#define SMEM_BYTES (OFF_RS+256)           // 132608 B total (2 WGs / 320KB WGP)
// Phase-0 f32 staging for async DMA: aliases OFF_OB (25600 B <= 32768 B)

__global__ __launch_bounds__(256)
void lfsa_fused(const float* __restrict__ x,
                const float* __restrict__ w_hidden,   // [384,64]
                const float* __restrict__ w_dw,       // [384,1,3,3]
                const float* __restrict__ norm_w,     // [128]
                const float* __restrict__ norm_b,     // [128]
                const float* __restrict__ w_out,      // [64,128]
                float* __restrict__ out)              // [4,64,256,256]
{
    extern __shared__ char smem[];
    _Float16* xh  = (_Float16*)(smem + OFF_XH);
    _Float16* whs = (_Float16*)(smem + OFF_WHS);
    float*    hid = (float*)   (smem + OFF_HID);
    float*    qb  = (float*)   (smem + OFF_QB);
    float*    kb  = (float*)   (smem + OFF_KB);
    float*    ob  = (float*)   (smem + OFF_OB);
    float*    vb  = (float*)   (smem + OFF_VB);
    _Float16* gb  = (_Float16*)(smem + OFF_GB);
    float*    mus = (float*)   (smem + OFF_MU);
    float*    rss = (float*)   (smem + OFF_RS);
    float*    xf  = (float*)   (smem + OFF_OB);   // phase-0 alias

    const int t    = threadIdx.x;
    const int lane = t & 31;
    const int wave = t >> 5;             // 0..7
    const int blk  = blockIdx.x;
    const int b    = blk >> 10;          // grid = 4 * 32 * 32
    const int ty   = (blk >> 5) & 31;    // patch row
    const int tx   = blk & 31;           // patch col
    const int gy0  = ty * 8 - 1;
    const int gx0  = tx * 8 - 1;

    // ---- Warm L2 for the (heavily reused) weight tensors
    for (int i = t * 32; i < 384 * DIMC; i += 256 * 32)
        __builtin_prefetch(w_hidden + i, 0, 1);
    if (t < 64) __builtin_prefetch(w_out + t * 128, 0, 1);

    // ---- Phase 0: async-DMA the 10x10x64 x-halo into LDS (f32 staging),
    //      then convert to f16 layout [pixel][ch] for WMMA B-fragments.
    for (int i = t; i < NPIX * DIMC; i += 256) {
        int p = i >> 6;
        int c = i & 63;
        int hy = p / HALO, hx = p % HALO;
        int gy = gy0 + hy, gx = gx0 + hx;
        if (gy >= 0 && gy < HH && gx >= 0 && gx < WW) {
            const float* gp = &x[(((size_t)b * DIMC + c) * HH + gy) * WW + gx];
            unsigned lds = (unsigned)(unsigned long long)(&xf[i]);
            unsigned long long ga = (unsigned long long)gp;
            asm volatile("global_load_async_to_lds_b32 %0, %1, off"
                         :: "v"(lds), "v"(ga) : "memory");
        } else {
            xf[i] = 0.0f;               // SAME zero padding at image borders
        }
    }
    asm volatile("s_wait_asynccnt 0x0" ::: "memory");
    __syncthreads();
    for (int i = t; i < NPAD * DIMC; i += 256)
        xh[i] = (_Float16)((i < NPIX * DIMC) ? xf[i] : 0.0f);
    __syncthreads();

    const int ch = t >> 4;        // 0..15 : channel within group
    const int p0 = t & 15;        // pixel base (dw-conv mapping)

    // ---- Main loop: 8 groups of 16 channels (q/k/v triples)
    for (int g = 0; g < 8; ++g) {
        // Phase 1a: stage group's 48 w_hidden rows into LDS as f16 (once;
        // A-fragments are reused across 7 N-tiles without re-converting)
        for (int i = t; i < 48 * DIMC; i += 256) {
            int r = i >> 6;                       // 0..47
            int row = (r >> 4) * 128 + g * 16 + (r & 15);
            whs[i] = (_Float16)w_hidden[row * DIMC + (i & 63)];
        }
        __syncthreads();

        // Phase 1b: GEMM1 -> hid[48][112] = whs * xh
        // 21 tile-jobs = 3 M-tiles (q,k,v) x 7 N-tiles, 2 K-steps of 32 each
        for (int job = wave; job < 21; job += 8) {
            int mt = job / 7;
            int nt = job % 7;
            v8f acc = {};
            for (int ks = 0; ks < 2; ++ks) {
                v16h a, bf;
                // A: lane = M; lanes>=16 take K halves 8..15 / 24..31
                const _Float16* wr = whs + (mt * 16 + (lane & 15)) * 64
                                         + ks * 32 + ((lane >> 4) ? 8 : 0);
                #pragma unroll
                for (int j = 0; j < 8; ++j) {
                    a[j]     = wr[j];
                    a[j + 8] = wr[16 + j];
                }
                // B: lane = N; lanes>=16 take K halves 8..15 / 24..31
                int n = nt * 16 + (lane & 15);
                const _Float16* xr = xh + n * 64 + ks * 32 + ((lane >> 4) ? 8 : 0);
                #pragma unroll
                for (int j = 0; j < 8; ++j) {
                    bf[j]     = xr[j];
                    bf[j + 8] = xr[16 + j];
                }
                acc = __builtin_amdgcn_wmma_f32_16x16x32_f16(
                        false, a, false, bf, (short)0, acc, false, false);
            }
            int n  = nt * 16 + (lane & 15);
            int mo = (lane >> 4) ? 8 : 0;
            #pragma unroll
            for (int r = 0; r < 8; ++r)
                hid[(mt * 16 + mo + r) * NPAD + n] = acc[r];
        }
        __syncthreads();

        // Phase 2a: depthwise 3x3 (cross-correlation, SAME pad via halo);
        // also zero the ob slice for this group's atomic accumulation.
        int oc = g * 16 + ch;
        float wq[9], wk[9], wv[9];
        #pragma unroll
        for (int j = 0; j < 9; ++j) {
            wq[j] = w_dw[oc * 9 + j];
            wk[j] = w_dw[(128 + oc) * 9 + j];
            wv[j] = w_dw[(256 + oc) * 9 + j];
        }
        #pragma unroll
        for (int i = 0; i < 4; ++i) {
            int pix = p0 + 16 * i;
            int py = pix >> 3, px = pix & 7;
            float aq = 0.f, ak = 0.f, av = 0.f;
            #pragma unroll
            for (int ky = 0; ky < 3; ++ky)
                #pragma unroll
                for (int kx = 0; kx < 3; ++kx) {
                    int hp = (py + ky) * HALO + (px + kx);
                    aq += hid[ch * NPAD + hp]        * wq[ky * 3 + kx];
                    ak += hid[(16 + ch) * NPAD + hp] * wk[ky * 3 + kx];
                    av += hid[(32 + ch) * NPAD + hp] * wv[ky * 3 + kx];
                }
            qb[ch * 64 + pix] = aq;
            kb[ch * 64 + pix] = ak;
            vb[(g * 16 + ch) * 64 + pix] = av;
            ob[(g * 16 + ch) * 64 + pix] = 0.0f;
        }
        __syncthreads();

        // Phase 2b: 8x8 circular conv (== irfft2(rfft2(q)*rfft2(k))),
        // row-register form: per (sy) load one q-row + one k-row (b128-able)
        // and do 64 FMAs on registers -> 0.25 LDS floats per FMA.
        // Two threads split sy 0..3 / 4..7; combine via 2-way ds_add_f32
        // (IEEE add is commutative -> bit-deterministic).
        {
            int th  = t & 127;
            int cc  = th >> 3;            // channel 0..15
            int py  = th & 7;             // output row
            int sy0 = (t >> 7) * 4;       // 0 or 4
            float acc[8] = {0.f,0.f,0.f,0.f,0.f,0.f,0.f,0.f};
            #pragma unroll
            for (int s = 0; s < 4; ++s) {
                int sy = sy0 + s;
                int ry = (py - sy) & 7;
                float qr[8], kr[8];
                #pragma unroll
                for (int j = 0; j < 8; ++j) {
                    qr[j] = qb[cc * 64 + sy * 8 + j];
                    kr[j] = kb[cc * 64 + ry * 8 + j];
                }
                #pragma unroll
                for (int px = 0; px < 8; ++px)
                    #pragma unroll
                    for (int sx = 0; sx < 8; ++sx)
                        acc[px] += qr[sx] * kr[(px - sx) & 7];
            }
            float* orow = ob + (g * 16 + cc) * 64 + py * 8;
            #pragma unroll
            for (int j = 0; j < 8; ++j)
                atomicAdd(&orow[j], acc[j]);
        }
        __syncthreads();
    }

    // ---- Phase 3: LayerNorm stats over 128 channels (deterministic serial sum)
    if (t < 64) {
        float s = 0.f, s2 = 0.f;
        for (int c = 0; c < 128; ++c) {
            float o = ob[c * 64 + t];
            s += o; s2 += o * o;
        }
        float m  = s * (1.0f / 128.0f);
        float vr = s2 * (1.0f / 128.0f) - m * m;
        mus[t] = m;
        rss[t] = rsqrtf(vr + 1e-5f);
    }
    __syncthreads();

    // ---- Phase 4: normalize, gate with v, pack as f16 B-matrix [pix][c]
    for (int i = t; i < 128 * 64; i += 256) {
        int c = i >> 6, pix = i & 63;
        float on = (ob[c * 64 + pix] - mus[pix]) * rss[pix] * norm_w[c] + norm_b[c];
        gb[pix * 128 + c] = (_Float16)(vb[c * 64 + pix] * on);
    }
    __syncthreads();

    // ---- Phase 5: GEMM2 out[64,64px] = w_out[64,128] @ gated[128,64px]
    for (int job = wave; job < 16; job += 8) {
        int mt = job >> 2, nt = job & 3;
        v8f acc = {};
        for (int ks = 0; ks < 4; ++ks) {
            v16h a, bf;
            const float* wr = w_out + (mt * 16 + (lane & 15)) * 128
                                    + ks * 32 + ((lane >> 4) ? 8 : 0);
            #pragma unroll
            for (int j = 0; j < 8; ++j) {
                a[j]     = (_Float16)wr[j];
                a[j + 8] = (_Float16)wr[16 + j];
            }
            int n = nt * 16 + (lane & 15);
            const _Float16* gr = gb + n * 128 + ks * 32 + ((lane >> 4) ? 8 : 0);
            #pragma unroll
            for (int j = 0; j < 8; ++j) {
                bf[j]     = gr[j];
                bf[j + 8] = gr[16 + j];
            }
            acc = __builtin_amdgcn_wmma_f32_16x16x32_f16(
                    false, a, false, bf, (short)0, acc, false, false);
        }
        int n  = nt * 16 + (lane & 15);
        int py = n >> 3, px = n & 7;
        int mo = (lane >> 4) ? 8 : 0;
        #pragma unroll
        for (int r = 0; r < 8; ++r) {
            int o = mt * 16 + mo + r;
            // streamed once, never re-read on device -> non-temporal
            __builtin_nontemporal_store(acc[r],
                &out[(((size_t)b * 64 + o) * HH + (ty * 8 + py)) * WW + (tx * 8 + px)]);
        }
    }
}

extern "C" void kernel_launch(void* const* d_in, const int* in_sizes, int n_in,
                              void* d_out, int out_size, void* d_ws, size_t ws_size,
                              hipStream_t stream) {
    (void)in_sizes; (void)n_in; (void)out_size; (void)d_ws; (void)ws_size;
    const float* x  = (const float*)d_in[0];
    const float* wh = (const float*)d_in[1];
    const float* wd = (const float*)d_in[2];
    const float* nw = (const float*)d_in[3];
    const float* nb = (const float*)d_in[4];
    const float* wo = (const float*)d_in[5];
    // d_in[6] = patch_size (== 8, hardcoded)
    float* o = (float*)d_out;

    hipFuncSetAttribute(reinterpret_cast<const void*>(lfsa_fused),
                        hipFuncAttributeMaxDynamicSharedMemorySize, SMEM_BYTES);
    // 4 batches * 32 * 32 patches = 4096 workgroups, 256 threads (8 waves) each
    lfsa_fused<<<4096, 256, SMEM_BYTES, stream>>>(x, wh, wd, nw, nb, wo, o);
}
